// _SlotAttention_45483703665381
// MI455X (gfx1250) — matrix operand order repaired and checked by reading, not compiled, not run on hardware
//
#include <hip/hip_runtime.h>
#include <hip/hip_bf16.h>

typedef __attribute__((ext_vector_type(16))) _Float16 v16h;
typedef __attribute__((ext_vector_type(8)))  _Float16 v8h;
typedef __attribute__((ext_vector_type(8)))  float    v8f;

#define BB   64
#define NN   4096
#define DD   128
#define SS   8
#define HH   256
#define BN   (BB * NN)          // 262144 rows
#define NSPLIT 8                // N-splits per batch in attention kernel

__device__ __forceinline__ v8f wmma_f16(v16h a, v16h b, v8f c) {
    return __builtin_amdgcn_wmma_f32_16x16x32_f16(
        /*neg_a=*/false, a, /*neg_b=*/false, b,
        /*c_mod=*/(short)0, c, /*reuse_a=*/false, /*reuse_b=*/false);
}

// A-fragment: 16x32 f16 tile from row-major [M][K] source, row stride ld (halfs).
// Lane L: m = L%16; K in {kb..kb+7, 16+kb..16+kb+7}, kb = (L<16)?0:8.
__device__ __forceinline__ v16h frag_A(const _Float16* p, int ld) {
    int lane = threadIdx.x & 31;
    int m  = lane & 15;
    int kb = (lane >> 4) << 3;            // 0 or 8
    const _Float16* r = p + m * ld + kb;
    union { v16h v; v8h h[2]; } u;
    u.h[0] = *(const v8h*)(r);            // K = kb .. kb+7
    u.h[1] = *(const v8h*)(r + 16);       // K = 16+kb .. 16+kb+7
    return u.v;
}

// B-fragment: 32x16 f16 tile, source stored transposed: [Ncol][K] row-major, stride ld.
// Lane L: n = L%16; K in {kb..kb+7, 16+kb..16+kb+7}, kb = (L<16)?0:8.
__device__ __forceinline__ v16h frag_BT(const _Float16* p, int ld) {
    int lane = threadIdx.x & 31;
    int n  = lane & 15;
    int kb = (lane >> 4) << 3;            // 0 or 8
    const _Float16* r = p + n * ld + kb;
    union { v16h v; v8h h[2]; } u;
    u.h[0] = *(const v8h*)(r);
    u.h[1] = *(const v8h*)(r + 16);
    return u.v;
}

// ---------------- small prep kernels ----------------

__global__ void init_slots_kernel(const float* __restrict__ noise,
                                  const float* __restrict__ mu,
                                  const float* __restrict__ sigma,
                                  float* __restrict__ slots, int n) {
    int i = blockIdx.x * blockDim.x + threadIdx.x;
    if (i >= n) return;
    int d = i & (DD - 1);
    float sg = sigma[d];
    float sp = (sg > 20.f) ? sg : log1pf(__expf(sg));   // softplus
    slots[i] = mu[d] + sp * noise[i];
}

__global__ void cvt_w_kernel(const float* __restrict__ wk, const float* __restrict__ wv,
                             _Float16* __restrict__ wk_h, _Float16* __restrict__ wv_h, int n) {
    int i = blockIdx.x * blockDim.x + threadIdx.x;
    if (i >= n) return;
    wk_h[i] = (_Float16)wk[i];
    wv_h[i] = (_Float16)wv[i];
}

// ---------------- fused LayerNorm + K/V projection (WMMA) ----------------
// Each block: 128 rows of x. LN -> f16 in LDS, then C[128x128] = Xln @ Wk^T and @ Wv^T.
// k stored row-major [row][e] f16 (A-operand later), v stored transposed [e][row] f16 (B-operand later).

__global__ __launch_bounds__(256) void ln_kv_kernel(
    const float* __restrict__ x, const float* __restrict__ lnw, const float* __restrict__ lnb,
    const _Float16* __restrict__ wk_h, const _Float16* __restrict__ wv_h,
    _Float16* __restrict__ k_h, _Float16* __restrict__ v_t) {

    __shared__ _Float16 xln[128][136];      // stride 136 halfs (16B-aligned rows)
    int wave = threadIdx.x >> 5;
    int lane = threadIdx.x & 31;
    long R0 = (long)blockIdx.x * 128;

    // LayerNorm: each wave handles 16 rows; 32 lanes x float4 per row
    float4 w4 = ((const float4*)lnw)[lane];
    float4 b4 = ((const float4*)lnb)[lane];
    for (int i = 0; i < 16; ++i) {
        int row = wave * 16 + i;
        const float* xr = x + (R0 + row) * DD;
        float4 a = ((const float4*)xr)[lane];
        float s  = a.x + a.y + a.z + a.w;
        float s2 = a.x*a.x + a.y*a.y + a.z*a.z + a.w*a.w;
        for (int off = 16; off; off >>= 1) {
            s  += __shfl_xor(s,  off, 32);
            s2 += __shfl_xor(s2, off, 32);
        }
        float mean = s * (1.f / 128.f);
        float var  = s2 * (1.f / 128.f) - mean * mean;
        float rstd = rsqrtf(var + 1e-5f);
        int c = lane * 4;
        xln[row][c + 0] = (_Float16)((a.x - mean) * rstd * w4.x + b4.x);
        xln[row][c + 1] = (_Float16)((a.y - mean) * rstd * w4.y + b4.y);
        xln[row][c + 2] = (_Float16)((a.z - mean) * rstd * w4.z + b4.z);
        xln[row][c + 3] = (_Float16)((a.w - mean) * rstd * w4.w + b4.w);
    }
    __syncthreads();

    // WMMA: wave owns M-tile rows [wave*16, wave*16+16)
    v16h afrag[4];
    for (int kc = 0; kc < 4; ++kc)
        afrag[kc] = frag_A(&xln[wave * 16][kc * 32], 136);

    for (int et = 0; et < 8; ++et) {
        v8f ck = {}; v8f cv = {};
        for (int kc = 0; kc < 4; ++kc) {
            v16h bk = frag_BT(wk_h + (et * 16) * DD + kc * 32, DD);
            v16h bv = frag_BT(wv_h + (et * 16) * DD + kc * 32, DD);
            ck = wmma_f16(afrag[kc], bk, ck);
            cv = wmma_f16(afrag[kc], bv, cv);
        }
        // C layout: lane col n = L%16, rows m = r + (L>=16 ? 8 : 0)
        int n  = lane & 15;
        int mo = (lane >> 4) * 8;
        long rowbase = R0 + wave * 16 + mo;
        int e = et * 16 + n;
        _Float16 hv[8];
        for (int r = 0; r < 8; ++r) {
            k_h[(rowbase + r) * DD + e] = (_Float16)ck[r];
            hv[r] = (_Float16)cv[r];
        }
        *(v8h*)(v_t + (long)e * BN + rowbase) = *(v8h*)hv;   // contiguous along rows
    }
}

// ---------------- q projection (tiny) ----------------
// One block per (b, s-padded-to-16). q stored as [b][s][e] f16 (B^T layout for dots).

__global__ __launch_bounds__(128) void q_kernel(
    const float* __restrict__ slots, const float* __restrict__ lnw, const float* __restrict__ lnb,
    const float* __restrict__ Wq, _Float16* __restrict__ q_h) {
    int blk = blockIdx.x;
    int b = blk >> 4, s = blk & 15, t = threadIdx.x;
    if (s >= SS) { q_h[blk * DD + t] = (_Float16)0.f; return; }
    __shared__ float sv[128], xln[128], stat[2];
    float val = slots[((b * SS + s) * DD) + t];
    sv[t] = val;
    __syncthreads();
    if (t == 0) {
        float m = 0.f;
        for (int i = 0; i < 128; ++i) m += sv[i];
        m *= (1.f / 128.f);
        float v = 0.f;
        for (int i = 0; i < 128; ++i) { float d0 = sv[i] - m; v += d0 * d0; }
        v *= (1.f / 128.f);
        stat[0] = m; stat[1] = rsqrtf(v + 1e-5f);
    }
    __syncthreads();
    xln[t] = (val - stat[0]) * stat[1] * lnw[t] + lnb[t];
    __syncthreads();
    float acc = 0.f;
    const float* wr = Wq + t * DD;
    for (int d0 = 0; d0 < DD; ++d0) acc += xln[d0] * wr[d0];
    q_h[blk * DD + t] = (_Float16)acc;
}

// ---------------- attention: dots (WMMA) -> softmax -> unnormalized updates (WMMA) ----------------
// grid (NSPLIT, B), block 256 = 8 waves. Each block covers 512 input rows of one batch.

__global__ __launch_bounds__(256) void attn_kernel(
    const _Float16* __restrict__ k_h, const _Float16* __restrict__ v_t,
    const _Float16* __restrict__ q_h,
    float* __restrict__ updates_part, float* __restrict__ sums_part) {

    __shared__ _Float16 attn_lds[16 * 136];   // [s][n], stride 136
    __shared__ float sums_w[8][16];
    int wave = threadIdx.x >> 5;
    int lane = threadIdx.x & 31;
    int split = blockIdx.x;
    int b = blockIdx.y;

    // q B-fragments (cols = 16 padded slots), loaded once
    const _Float16* qb = q_h + b * 16 * DD;
    v16h bq[4];
    for (int kc = 0; kc < 4; ++kc) bq[kc] = frag_BT(qb + kc * 32, DD);

    v8f upd = {};            // wave's d-tile accumulator (dt = wave), persists over chunks
    float ssum = 0.f;
    const float scale = 0.08838834764831845f;   // 1/sqrt(128)
    long base = (long)b * NN + (long)split * (NN / NSPLIT);

    for (int c = 0; c < 4; ++c) {
        long g = base + c * 128;
        // --- phase 1: dots for n-subtile rows [wave*16, wave*16+16), softmax over slots ---
        v8f dots = {};
        for (int kc = 0; kc < 4; ++kc) {
            v16h a = frag_A(k_h + (g + wave * 16) * DD + kc * 32, DD);
            dots = wmma_f16(a, bq[kc], dots);
        }
        int  s     = lane & 15;
        bool valid = (s < SS);
        int  mo    = (lane >> 4) * 8;
        for (int r = 0; r < 8; ++r) {
            float xx = valid ? dots[r] * scale : -1e30f;
            float mx = xx;
            for (int off = 1; off < 16; off <<= 1) mx = fmaxf(mx, __shfl_xor(mx, off, 32));
            float ex = valid ? __expf(xx - mx) : 0.f;
            float sm = ex;
            for (int off = 1; off < 16; off <<= 1) sm += __shfl_xor(sm, off, 32);
            float at = valid ? (ex / sm + 1e-8f) : 0.f;
            ssum += at;
            attn_lds[s * 136 + wave * 16 + mo + r] = (_Float16)at;   // transposed: [s][n]
        }
        __syncthreads();
        // --- phase 2: updates += attn^T @ v ; wave owns d-tile dt = wave ---
        for (int kc = 0; kc < 4; ++kc) {
            v16h a  = frag_A(attn_lds + kc * 32, 136);                       // M=s, K=n
            v16h bv = frag_BT(v_t + (long)(wave * 16) * BN + g + kc * 32, BN); // [d][n]
            upd = wmma_f16(a, bv, upd);
        }
        __syncthreads();
    }

    // per-slot attention sums: combine lane pairs (L, L^16), then across waves
    ssum += __shfl_xor(ssum, 16, 32);
    if (lane < 16) sums_w[wave][lane] = ssum;
    __syncthreads();
    if (wave == 0 && lane < SS) {
        float t = 0.f;
        for (int w = 0; w < 8; ++w) t += sums_w[w][lane];
        sums_part[(b * NSPLIT + split) * SS + lane] = t;
    }
    // store updates partial: C layout M=s, N=d; lanes<16 hold real slots s=0..7
    if (lane < 16) {
        int dd = wave * 16 + lane;
        for (int r = 0; r < 8; ++r)
            updates_part[(((long)(b * NSPLIT + split)) * SS + r) * DD + dd] = upd[r];
    }
}

// ---------------- GRU cell update (tiny) ----------------

__global__ __launch_bounds__(128) void gru_kernel(
    const float* __restrict__ updates_part, const float* __restrict__ sums_part,
    const float* __restrict__ W_ih, const float* __restrict__ W_hh,
    const float* __restrict__ b_ih, const float* __restrict__ b_hh,
    float* __restrict__ slots) {
    int blk = blockIdx.x;
    int b = blk >> 3, s = blk & 7, t = threadIdx.x;
    __shared__ float u[128], hp[128];
    float acc = 0.f;
    for (int sp = 0; sp < NSPLIT; ++sp)
        acc += updates_part[(((b * NSPLIT + sp) * SS) + s) * DD + t];
    float ssum = 0.f;
    for (int sp = 0; sp < NSPLIT; ++sp)
        ssum += sums_part[(b * NSPLIT + sp) * SS + s];
    u[t] = acc / ssum;
    float prev = slots[(b * SS + s) * DD + t];
    hp[t] = prev;
    __syncthreads();
    float gi[3], gh[3];
    for (int g = 0; g < 3; ++g) {
        int row = g * DD + t;
        float a1 = b_ih[row], a2 = b_hh[row];
        const float* wi = W_ih + row * DD;
        const float* wh = W_hh + row * DD;
        for (int d0 = 0; d0 < DD; ++d0) { a1 += u[d0] * wi[d0]; a2 += hp[d0] * wh[d0]; }
        gi[g] = a1; gh[g] = a2;
    }
    float r  = 1.f / (1.f + __expf(-(gi[0] + gh[0])));
    float z  = 1.f / (1.f + __expf(-(gi[1] + gh[1])));
    float nn = tanhf(gi[2] + r * gh[2]);
    slots[(b * SS + s) * DD + t] = (1.f - z) * nn + z * prev;
}

// ---------------- final MLP + residual (tiny) ----------------

__global__ __launch_bounds__(256) void mlp_kernel(
    const float* __restrict__ slots, const float* __restrict__ lnw, const float* __restrict__ lnb,
    const float* __restrict__ W1, const float* __restrict__ b1,
    const float* __restrict__ W2, const float* __restrict__ b2,
    float* __restrict__ out) {
    int blk = blockIdx.x, t = threadIdx.x;
    __shared__ float sv[128], xln[128], hh[256], stat[2];
    long row = (long)blk * DD;
    if (t < 128) sv[t] = slots[row + t];
    __syncthreads();
    if (t == 0) {
        float m = 0.f;
        for (int i = 0; i < 128; ++i) m += sv[i];
        m *= (1.f / 128.f);
        float v = 0.f;
        for (int i = 0; i < 128; ++i) { float d0 = sv[i] - m; v += d0 * d0; }
        v *= (1.f / 128.f);
        stat[0] = m; stat[1] = rsqrtf(v + 1e-5f);
    }
    __syncthreads();
    if (t < 128) xln[t] = (sv[t] - stat[0]) * stat[1] * lnw[t] + lnb[t];
    __syncthreads();
    float a = b1[t];
    const float* w = W1 + t * DD;
    for (int d0 = 0; d0 < DD; ++d0) a += xln[d0] * w[d0];
    hh[t] = fmaxf(a, 0.f);
    __syncthreads();
    if (t < 128) {
        float o = sv[t] + b2[t];
        const float* w2 = W2 + t * HH;
        for (int j = 0; j < HH; ++j) o += hh[j] * w2[j];
        out[row + t] = o;
    }
}

// ---------------- host launcher ----------------

extern "C" void kernel_launch(void* const* d_in, const int* in_sizes, int n_in,
                              void* d_out, int out_size, void* d_ws, size_t ws_size,
                              hipStream_t stream) {
    const float* x      = (const float*)d_in[0];
    const float* noise  = (const float*)d_in[1];
    const float* ln_in_w = (const float*)d_in[2];
    const float* ln_in_b = (const float*)d_in[3];
    const float* ln_s_w  = (const float*)d_in[4];
    const float* ln_s_b  = (const float*)d_in[5];
    const float* ln_m_w  = (const float*)d_in[6];
    const float* ln_m_b  = (const float*)d_in[7];
    const float* slot_mu    = (const float*)d_in[8];
    const float* slot_sigma = (const float*)d_in[9];
    const float* Wq   = (const float*)d_in[10];
    const float* Wk   = (const float*)d_in[11];
    const float* Wv   = (const float*)d_in[12];
    const float* W_ih = (const float*)d_in[13];
    const float* W_hh = (const float*)d_in[14];
    const float* b_ih = (const float*)d_in[15];
    const float* b_hh = (const float*)d_in[16];
    const float* W1 = (const float*)d_in[17];
    const float* b1 = (const float*)d_in[18];
    const float* W2 = (const float*)d_in[19];
    const float* b2 = (const float*)d_in[20];
    float* out = (float*)d_out;

    char* ws = (char*)d_ws;
    size_t off = 0;
    _Float16* k_h = (_Float16*)(ws + off);           off += (size_t)BN * DD * 2;     // 64 MB
    _Float16* v_t = (_Float16*)(ws + off);           off += (size_t)BN * DD * 2;     // 64 MB
    _Float16* wk_h = (_Float16*)(ws + off);          off += (size_t)DD * DD * 2;
    _Float16* wv_h = (_Float16*)(ws + off);          off += (size_t)DD * DD * 2;
    float* slots = (float*)(ws + off);               off += (size_t)BB * SS * DD * 4;
    _Float16* q_h = (_Float16*)(ws + off);           off += (size_t)BB * 16 * DD * 2;
    float* updates_part = (float*)(ws + off);        off += (size_t)BB * NSPLIT * SS * DD * 4;
    float* sums_part = (float*)(ws + off);           off += (size_t)BB * NSPLIT * SS * 4;

    // prep
    init_slots_kernel<<<(BB * SS * DD + 255) / 256, 256, 0, stream>>>(
        noise, slot_mu, slot_sigma, slots, BB * SS * DD);
    cvt_w_kernel<<<(DD * DD + 255) / 256, 256, 0, stream>>>(Wk, Wv, wk_h, wv_h, DD * DD);

    // fused LN + K/V projection
    ln_kv_kernel<<<BN / 128, 256, 0, stream>>>(x, ln_in_w, ln_in_b, wk_h, wv_h, k_h, v_t);

    // 3 slot-attention iterations
    for (int it = 0; it < 3; ++it) {
        q_kernel<<<BB * 16, 128, 0, stream>>>(slots, ln_s_w, ln_s_b, Wq, q_h);
        attn_kernel<<<dim3(NSPLIT, BB), 256, 0, stream>>>(k_h, v_t, q_h, updates_part, sums_part);
        gru_kernel<<<BB * SS, 128, 0, stream>>>(updates_part, sums_part,
                                                W_ih, W_hh, b_ih, b_hh, slots);
    }

    // final MLP + residual -> out
    mlp_kernel<<<BB * SS, 256, 0, stream>>>(slots, ln_m_w, ln_m_b, W1, b1, W2, b2, out);

    (void)in_sizes; (void)n_in; (void)out_size; (void)ws_size;
}
